// FlashRetAttention_38517266710798
// MI455X (gfx1250) — matrix operand order repaired and checked by reading, not compiled, not run on hardware
//
#include <hip/hip_runtime.h>

typedef __attribute__((ext_vector_type(16))) _Float16 v16h;
typedef __attribute__((ext_vector_type(8)))  _Float16 v8h;
typedef __attribute__((ext_vector_type(4)))  _Float16 v4h;
typedef __attribute__((ext_vector_type(8)))  float    v8f;

// Problem shape (fixed by the reference)
constexpr int B = 2, H = 16, S = 2048, D = 64;

constexpr int WAVES         = 8;                      // 256 threads = 8 wave32
constexpr int ROWS_PER_WAVE = 16;                     // 16-row query tile per wave
constexpr int M_TILE        = WAVES * ROWS_PER_WAVE;  // 128 query rows per block
constexpr int KCHUNK        = 64;                     // keys per iteration (4 score tiles)

// o[b,h,q,:] = sum_k (qk*mask) v / max(sum_k |qk*mask|, 1)
__global__ __launch_bounds__(256, 2)
void fra_wmma_kernel(const float* __restrict__ q,
                     const float* __restrict__ k,
                     const float* __restrict__ v,
                     const float* __restrict__ mask,
                     float* __restrict__ out)
{
    // LDS: 16K (Q) + 8K (Kt) + 8K (V) + 16K (P) = 48 KB / block
    __shared__ alignas(32) _Float16 sQ [M_TILE][D];       // [row][d]
    __shared__ alignas(32) _Float16 sKt[D][KCHUNK];       // [d][key]  (transposed!)
    __shared__ alignas(32) _Float16 sV [KCHUNK][D];       // [key][d]
    __shared__ alignas(32) _Float16 sP [WAVES][ROWS_PER_WAVE][KCHUNK];

    const int tid  = threadIdx.x;
    const int lane = tid & 31;
    const int wave = tid >> 5;
    const int row  = lane & 15;   // N / column index inside a 16-wide tile
    const int hi   = lane >> 4;   // 0: C rows 0..7, 1: C rows 8..15

    const int bh = blockIdx.y;            // (b*H + h)
    const int q0 = blockIdx.x * M_TILE;   // first query row of this block

    const float* qbase = q    + (size_t)bh * S * D;
    const float* kbase = k    + (size_t)bh * S * D;
    const float* vbase = v    + (size_t)bh * S * D;
    const float* mbase = mask + (size_t)bh * S * S;
    float*       obase = out  + (size_t)bh * S * D;

    // ---- Stage Q tile (f32 -> f16), vectorized: float4 load + v4h store ----
    for (int i = tid * 4; i < M_TILE * D; i += 256 * 4) {
        const float4 x = *(const float4*)&qbase[(size_t)q0 * D + i];
        v4h h; h[0] = (_Float16)x.x; h[1] = (_Float16)x.y;
               h[2] = (_Float16)x.z; h[3] = (_Float16)x.w;
        *(v4h*)&(&sQ[0][0])[i] = h;
    }
    __syncthreads();

    // ---- Q A-fragments: per lane, two aligned 16B runs per 16x32 fragment ----
    // A 16x32 f16 layout: lane holds row (lane&15); halves = K in
    //   {8*hi .. 8*hi+7} U {8*hi+16 .. 8*hi+23}   (+32c)
    v16h aQ[2];
    #pragma unroll
    for (int c = 0; c < 2; ++c) {
        const _Float16* qrow = &sQ[wave * ROWS_PER_WAVE + row][0];
        const v8h lo = *(const v8h*)&qrow[32 * c + 8 * hi];
        const v8h up = *(const v8h*)&qrow[32 * c + 8 * hi + 16];
        aQ[c] = __builtin_shufflevector(lo, up, 0,1,2,3,4,5,6,7,
                                                8,9,10,11,12,13,14,15);
    }

    v8f   oAcc[4] = {};   // 4 x (16x16 f32) output tiles covering D=64
    float rAcc[8] = {};   // per-lane partial |p| row sums (C-layout rows)

    for (int kc = 0; kc < S; kc += KCHUNK) {
        __syncthreads();  // previous chunk's sKt/sV reads done

        // ---- Stage K chunk transposed: float4 loads, scalar b16 stores ----
        for (int i = tid * 4; i < KCHUNK * D; i += 256 * 4) {
            const float4 x = *(const float4*)&kbase[(size_t)kc * D + i];
            const int key = i / D, d = i % D;
            sKt[d + 0][key] = (_Float16)x.x;
            sKt[d + 1][key] = (_Float16)x.y;
            sKt[d + 2][key] = (_Float16)x.z;
            sKt[d + 3][key] = (_Float16)x.w;
        }
        // ---- Stage V chunk row-major: float4 load + v4h store ----
        for (int i = tid * 4; i < KCHUNK * D; i += 256 * 4) {
            const float4 x = *(const float4*)&vbase[(size_t)kc * D + i];
            v4h h; h[0] = (_Float16)x.x; h[1] = (_Float16)x.y;
                   h[2] = (_Float16)x.z; h[3] = (_Float16)x.w;
            *(v4h*)&(&sV[0][0])[i] = h;
        }
        // Prefetch next K/V chunk + next mask block while computing this one
        if (kc + KCHUNK < S) {
            __builtin_prefetch(&kbase[(size_t)(kc + KCHUNK) * D + tid * 16], 0, 1);
            __builtin_prefetch(&vbase[(size_t)(kc + KCHUNK) * D + tid * 16], 0, 1);
            __builtin_prefetch(&mbase[(size_t)(q0 + wave * ROWS_PER_WAVE + 8 * hi) * S
                                      + kc + KCHUNK + row * 4], 0, 0);
        }
        __syncthreads();

        // ---- Scores: four 16x16 tiles (keys kc+16t .. kc+16t+15) ----
        #pragma unroll
        for (int t = 0; t < 4; ++t) {
            v8f sc = {};
            #pragma unroll
            for (int c = 0; c < 2; ++c) {
                // B 32x16 f16: lane l holds column n=l&15 (key 16t+n);
                // elements = d 32c + table-order k (8hi+s | 16+8hi+s-8)
                v16h bK;
                #pragma unroll
                for (int s = 0; s < 16; ++s)
                    bK[s] = sKt[32 * c + (s < 8 ? 8 * hi + s : 16 + 8 * hi + (s - 8))][16 * t + row];
                sc = __builtin_amdgcn_wmma_f32_16x16x32_f16(
                        false, aQ[c], false, bK, (short)0, sc, false, false);
            }
            // ---- mask multiply (NT stream), |p| accumulate, spill P to LDS ----
            #pragma unroll
            for (int i = 0; i < 8; ++i) {
                const int mrow = q0 + wave * ROWS_PER_WAVE + i + 8 * hi;
                const int mcol = kc + 16 * t + row;
                const float m  = __builtin_nontemporal_load(
                                     &mbase[(size_t)mrow * S + mcol]);
                const float p  = sc[i] * m;
                rAcc[i] += __builtin_fabsf(p);
                sP[wave][i + 8 * hi][16 * t + row] = (_Float16)p;
            }
        }

        // Same-wave LDS RAW on sP: DS ops from one wave stay in order.
        // ---- P A-fragments: two aligned 16B runs each ----
        v16h aP[2];
        #pragma unroll
        for (int c = 0; c < 2; ++c) {
            const _Float16* prow = &sP[wave][row][0];
            const v8h lo = *(const v8h*)&prow[32 * c + 8 * hi];
            const v8h up = *(const v8h*)&prow[32 * c + 8 * hi + 16];
            aP[c] = __builtin_shufflevector(lo, up, 0,1,2,3,4,5,6,7,
                                                    8,9,10,11,12,13,14,15);
        }

        // ---- O(16x64) += P(16x64) x V(64x64): 4 col tiles x 2 K-chunks ----
        #pragma unroll
        for (int t = 0; t < 4; ++t) {
            #pragma unroll
            for (int c = 0; c < 2; ++c) {
                // B 32x16: lane l holds column n=l&15 (V col 16t+n); elements = key 32c + table-order k
                v16h bV;
                #pragma unroll
                for (int s = 0; s < 16; ++s)
                    bV[s] = sV[32 * c + (s < 8 ? 8 * hi + s : 16 + 8 * hi + (s - 8))][16 * t + row];
                oAcc[t] = __builtin_amdgcn_wmma_f32_16x16x32_f16(
                            false, aP[c], false, bV, (short)0, oAcc[t], false, false);
            }
        }
    }

    // ---- Row-sum |p| reduction across 16-lane groups (matches C layout),
    //      normalize, store ----
    #pragma unroll
    for (int i = 0; i < 8; ++i) {
        float r = rAcc[i];
        #pragma unroll
        for (int m = 1; m < 16; m <<= 1)
            r += __shfl_xor(r, m, 32);            // stays within each 16-lane group
        const float inv = 1.0f / fmaxf(r, 1.0f);

        const int orow = q0 + wave * ROWS_PER_WAVE + i + 8 * hi;
        #pragma unroll
        for (int t = 0; t < 4; ++t)
            obase[(size_t)orow * D + 16 * t + row] = oAcc[t][i] * inv;
    }
}

extern "C" void kernel_launch(void* const* d_in, const int* in_sizes, int n_in,
                              void* d_out, int out_size, void* d_ws, size_t ws_size,
                              hipStream_t stream) {
    (void)in_sizes; (void)n_in; (void)out_size; (void)d_ws; (void)ws_size;
    const float* q    = (const float*)d_in[0];
    const float* k    = (const float*)d_in[1];
    const float* v    = (const float*)d_in[2];
    const float* mask = (const float*)d_in[3];
    float* out        = (float*)d_out;

    dim3 grid(S / M_TILE, B * H);   // (16, 32) blocks
    fra_wmma_kernel<<<grid, 256, 0, stream>>>(q, k, v, mask, out);
}
